// BertAttention_377957122364
// MI455X (gfx1250) — matrix operand order repaired and checked by reading, not compiled, not run on hardware
//
#include <hip/hip_runtime.h>

#define S_LEN 4096
#define HID   1024
#define NHEAD 16
#define HDIM  64

typedef __bf16 bf16_t;
typedef __attribute__((ext_vector_type(16))) bf16_t v16bf;
typedef __attribute__((ext_vector_type(8)))  float  v8f;

union Frag {
  v16bf v;
  unsigned int u[8];
};

// ---------------- async global->LDS copy (CDNA5 ASYNCcnt path), with fallback
#if defined(__gfx1250__) && __has_builtin(__builtin_amdgcn_global_load_async_to_lds_b128)
#define HAVE_ASYNC 1
#else
#define HAVE_ASYNC 0
#endif

// prototype (from hipcc diagnostic): (v4i AS1*, v4i AS3*, imm offset, imm cpol)
typedef int v4i_vs __attribute__((vector_size(16)));
typedef __attribute__((address_space(1))) v4i_vs* gv4_t;
typedef __attribute__((address_space(3))) v4i_vs* lv4_t;

#if HAVE_ASYNC
#define ASYNC_CP16(gsrc, ldst)                                           \
  __builtin_amdgcn_global_load_async_to_lds_b128(                        \
      (gv4_t)(unsigned long long)(gsrc),                                 \
      (lv4_t)(unsigned int)(unsigned long long)(ldst), 0, 0)
#if __has_builtin(__builtin_amdgcn_s_wait_asynccnt)
#define STAGE_WAIT(n) __builtin_amdgcn_s_wait_asynccnt(n)
#else
#define STAGE_WAIT(n) asm volatile("s_wait_asynccnt %0" ::"i"(n) : "memory")
#endif
#else
#define ASYNC_CP16(gsrc, ldst)                                           \
  do { *(uint4*)(ldst) = *(const uint4*)(gsrc); } while (0)
#define STAGE_WAIT(n) ((void)0)
#endif

__device__ __forceinline__ unsigned short f2bf(float f) {
  unsigned int u = __float_as_uint(f);
  u += 0x7FFFu + ((u >> 16) & 1u);   // round-to-nearest-even
  return (unsigned short)(u >> 16);
}

// ---------------------------------------------------------------- conversions
__global__ __launch_bounds__(256)
void cvt_f32_to_bf16(const float* __restrict__ src, unsigned short* __restrict__ dst, int n) {
  int i = blockIdx.x * blockDim.x + threadIdx.x;
  int stride = gridDim.x * blockDim.x;
  for (; i < n; i += stride) dst[i] = f2bf(src[i]);
}

// dst[k][n] = bf16(src[n][k])  (store W^T, k-major, so B-tiles are contiguous)
__global__ __launch_bounds__(256)
void cvt_transpose_bf16(const float* __restrict__ src, unsigned short* __restrict__ dst) {
  __shared__ float tile[32][33];
  const int k0 = blockIdx.x * 32, n0 = blockIdx.y * 32;
  const int tx = threadIdx.x, ty = threadIdx.y;   // (32, 8)
  #pragma unroll
  for (int i = 0; i < 4; ++i)
    tile[ty + i * 8][tx] = src[(size_t)(n0 + ty + i * 8) * HID + k0 + tx];
  __syncthreads();
  #pragma unroll
  for (int i = 0; i < 4; ++i)
    dst[(size_t)(k0 + ty + i * 8) * HID + n0 + tx] = f2bf(tile[tx][ty + i * 8]);
}

// ---------------------------------------------------------------- QKV GEMM
// Y = X @ W^T + b ; W passed pre-transposed (k-major [H][H]).
// Q,V stored bf16 [head][S][64]; K stored transposed bf16 [H][S]; Q scaled 0.125
__global__ __launch_bounds__(256)
void qkv_gemm_kernel(const unsigned short* __restrict__ Xbf,   // [S][H]
                     const unsigned short* __restrict__ Wtq,   // [H][H] k-major
                     const unsigned short* __restrict__ Wtk,
                     const unsigned short* __restrict__ Wtv,
                     const float* __restrict__ bq,
                     const float* __restrict__ bk,
                     const float* __restrict__ bv,
                     unsigned short* __restrict__ Qo,
                     unsigned short* __restrict__ Ko,
                     unsigned short* __restrict__ Vo) {
  __shared__ unsigned short Xs[2][64][32];    // 2 x 4 KB
  __shared__ unsigned short Ws[2][32][128];   // 2 x 8 KB

  const unsigned short* Wt; const float* bias; unsigned short* out;
  float scale; int ktlayout;
  if (blockIdx.z == 0)      { Wt = Wtq; bias = bq; out = Qo; scale = 0.125f; ktlayout = 0; }
  else if (blockIdx.z == 1) { Wt = Wtk; bias = bk; out = Ko; scale = 1.0f;   ktlayout = 1; }
  else                      { Wt = Wtv; bias = bv; out = Vo; scale = 1.0f;   ktlayout = 0; }

  const int t    = threadIdx.x;
  const int wave = t >> 5, lane = t & 31, lan = lane & 15, hl = lane >> 4;
  const int wm   = wave & 3;          // 4 M-waves
  const int wn   = wave >> 2;         // 2 N-waves
  const int M0   = blockIdx.y * 64;
  const int N0   = blockIdx.x * 128;

  auto issue = [&](int kk, int buf) {
    // X tile: 64 rows x 64B = 256 x 16B chunks, 1 per thread
    {
      int row = t >> 2, sub = t & 3;
      ASYNC_CP16(Xbf + (size_t)(M0 + row) * HID + kk * 32 + sub * 8,
                 &Xs[buf][row][sub * 8]);
    }
    // W^T tile: 32 rows x 256B = 512 x 16B chunks, 2 per thread
    #pragma unroll
    for (int i = 0; i < 2; ++i) {
      int c = i * 256 + t;
      int k = c >> 4, sub = c & 15;
      ASYNC_CP16(Wt + (size_t)(kk * 32 + k) * HID + N0 + sub * 8,
                 &Ws[buf][k][sub * 8]);
    }
  };

  v8f zero8 = {0.f,0.f,0.f,0.f,0.f,0.f,0.f,0.f};
  v8f acc[4];
  #pragma unroll
  for (int j = 0; j < 4; ++j) acc[j] = zero8;

  issue(0, 0);
  const int NK = HID / 32;
  for (int kk = 0; kk < NK; ++kk) {
    const int buf = kk & 1;
    if (kk + 1 < NK) { issue(kk + 1, buf ^ 1); STAGE_WAIT(3); }
    else             { STAGE_WAIT(0); }
    __syncthreads();

    Frag af;
    const unsigned int* xrow = (const unsigned int*)&Xs[buf][wm * 16 + lan][0];
    #pragma unroll
    for (int j = 0; j < 8; ++j)
      af.u[j] = xrow[(j & 3) + ((j >> 2) << 3) + (hl << 2)];

    const unsigned int* wsu = (const unsigned int*)&Ws[buf][0][0];  // [32][64]
    #pragma unroll
    for (int jt = 0; jt < 4; ++jt) {
      Frag bfr;
      const unsigned short* bcol = &Ws[buf][0][wn * 64 + jt * 16 + lan];
      #pragma unroll
      for (int j = 0; j < 8; ++j) { int k = 2 * (j & 3) + ((j >> 2) << 4) + (hl << 3); bfr.u[j] = bcol[k * 128] | ((unsigned int)bcol[(k + 1) * 128] << 16); }
      acc[jt] = __builtin_amdgcn_wmma_f32_16x16x32_bf16(
          false, af.v, false, bfr.v, (short)0, acc[jt], false, false);
    }
    __syncthreads();
  }

  #pragma unroll
  for (int jt = 0; jt < 4; ++jt) {
    int col = N0 + wn * 64 + jt * 16 + lan;
    float b = bias[col];
    int head = col >> 6, d = col & 63;
    #pragma unroll
    for (int r = 0; r < 8; ++r) {
      int row = M0 + wm * 16 + r + 8 * hl;
      float v = (acc[jt][r] + b) * scale;
      size_t idx = ktlayout ? ((size_t)col * S_LEN + row)
                            : ((((size_t)head * S_LEN + row) << 6) + d);
      out[idx] = f2bf(v);
    }
  }
}

// ---------------------------------------------------------------- flash attention
// grid (S/64, NHEAD); 4 waves, each owns 16 query rows; online softmax.
__global__ __launch_bounds__(128)
void flash_attn_kernel(const unsigned short* __restrict__ Qg,   // [NH][S][64]
                       const unsigned short* __restrict__ Ktg,  // [H][S]  (K^T)
                       const unsigned short* __restrict__ Vg,   // [NH][S][64]
                       const float* __restrict__ mask,          // [S]
                       unsigned short* __restrict__ ctx) {      // [S][H]
  __shared__ unsigned short KtS[2][64][32];   // 2 x 4 KB  [d][key]
  __shared__ unsigned short VsS[2][32][64];   // 2 x 4 KB  [key][d]
  __shared__ unsigned short Pt[4][16][32];    // per-wave P transpose buffer

  const int t    = threadIdx.x;
  const int wave = t >> 5, lane = t & 31, lan = lane & 15, hl = lane >> 4;
  const int h  = blockIdx.y;
  const int Q0 = blockIdx.x * 64;
  const int qrow = Q0 + wave * 16 + lan;

  // Q A-fragments (hd=64 -> two k-chunks of 32), loaded once
  Frag qa[2];
  const unsigned int* qbase =
      (const unsigned int*)Qg + ((size_t)h * S_LEN + qrow) * 32;
  #pragma unroll
  for (int c = 0; c < 2; ++c)
    #pragma unroll
    for (int j = 0; j < 8; ++j)
      qa[c].u[j] = qbase[(c << 4) + (j & 3) + ((j >> 2) << 3) + (hl << 2)];

  auto issue = [&](int kb, int buf) {
    const int key0 = kb * 32;
    #pragma unroll
    for (int i = 0; i < 2; ++i) {             // K^T tile: 64 rows x 64B = 256 chunks
      int c = i * 128 + t;
      int d = c >> 2, sub = c & 3;
      ASYNC_CP16(Ktg + (size_t)(h * 64 + d) * S_LEN + key0 + sub * 8,
                 &KtS[buf][d][sub * 8]);
    }
    #pragma unroll
    for (int i = 0; i < 2; ++i) {             // V tile: 32 rows x 128B = 256 chunks
      int c = i * 128 + t;
      int key = c >> 3, sub = c & 7;
      ASYNC_CP16(Vg + ((size_t)h * S_LEN + key0 + key) * 64 + sub * 8,
                 &VsS[buf][key][sub * 8]);
    }
  };

  v8f zero8 = {0.f,0.f,0.f,0.f,0.f,0.f,0.f,0.f};
  v8f cacc[4];
  #pragma unroll
  for (int j = 0; j < 4; ++j) cacc[j] = zero8;
  float run_max[8], run_sum[8];
  #pragma unroll
  for (int r = 0; r < 8; ++r) { run_max[r] = -1e30f; run_sum[r] = 0.f; }

  issue(0, 0);
  const int NKB = S_LEN / 32;
  for (int kb = 0; kb < NKB; ++kb) {
    const int buf = kb & 1;
    if (kb + 1 < NKB) { issue(kb + 1, buf ^ 1); STAGE_WAIT(4); }
    else              { STAGE_WAIT(0); }
    __syncthreads();

    const int key0 = kb * 32;
    float mk0 = mask[key0 + lan];
    float mk1 = mask[key0 + 16 + lan];

    v8f s0 = zero8, s1 = zero8;
    #pragma unroll
    for (int c = 0; c < 2; ++c) {
      Frag b0, b1;
      const unsigned short* kc = &KtS[buf][c * 32][lan];
      #pragma unroll
      for (int j = 0; j < 8; ++j) { int k = 2 * (j & 3) + ((j >> 2) << 4) + (hl << 3); b0.u[j] = kc[k * 32] | ((unsigned int)kc[(k + 1) * 32] << 16); b1.u[j] = kc[k * 32 + 16] | ((unsigned int)kc[(k + 1) * 32 + 16] << 16); }
      s0 = __builtin_amdgcn_wmma_f32_16x16x32_bf16(false, qa[c].v, false, b0.v, (short)0, s0, false, false);
      s1 = __builtin_amdgcn_wmma_f32_16x16x32_bf16(false, qa[c].v, false, b1.v, (short)0, s1, false, false);
    }

    float corr[8];
    #pragma unroll
    for (int r = 0; r < 8; ++r) {
      float a0 = s0[r] + mk0, a1 = s1[r] + mk1;
      float mx = fmaxf(a0, a1);
      #pragma unroll
      for (int off = 8; off >= 1; off >>= 1) mx = fmaxf(mx, __shfl_xor(mx, off));
      float nm = fmaxf(run_max[r], mx);
      float c0 = __expf(run_max[r] - nm);
      run_max[r] = nm;
      float p0 = __expf(a0 - nm), p1 = __expf(a1 - nm);
      float ps = p0 + p1;
      #pragma unroll
      for (int off = 8; off >= 1; off >>= 1) ps += __shfl_xor(ps, off);
      run_sum[r] = run_sum[r] * c0 + ps;
      corr[r] = c0;
      Pt[wave][r + 8 * hl][lan]      = f2bf(p0);   // P transpose via LDS
      Pt[wave][r + 8 * hl][16 + lan] = f2bf(p1);
    }
    #pragma unroll
    for (int jt = 0; jt < 4; ++jt)
      #pragma unroll
      for (int r = 0; r < 8; ++r) cacc[jt][r] *= corr[r];

    // same-wave LDS ops are in-order; wait for this wave's P stores to land
    asm volatile("s_wait_dscnt 0" ::: "memory");

    Frag pa;
    const unsigned int* pu = (const unsigned int*)&Pt[wave][lan][0];
    #pragma unroll
    for (int j = 0; j < 8; ++j)
      pa.u[j] = pu[(j & 3) + ((j >> 2) << 3) + (hl << 2)];

    #pragma unroll
    for (int jt = 0; jt < 4; ++jt) {
      Frag bv;
      const unsigned short* vc = &VsS[buf][0][jt * 16 + lan];
      #pragma unroll
      for (int j = 0; j < 8; ++j) { int k = 2 * (j & 3) + ((j >> 2) << 4) + (hl << 3); bv.u[j] = vc[k * 64] | ((unsigned int)vc[(k + 1) * 64] << 16); }
      cacc[jt] = __builtin_amdgcn_wmma_f32_16x16x32_bf16(
          false, pa.v, false, bv.v, (short)0, cacc[jt], false, false);
    }
    __syncthreads();
  }

  #pragma unroll
  for (int r = 0; r < 8; ++r) run_sum[r] = 1.0f / run_sum[r];
  #pragma unroll
  for (int jt = 0; jt < 4; ++jt) {
    int col = h * 64 + jt * 16 + lan;
    #pragma unroll
    for (int r = 0; r < 8; ++r) {
      int row = Q0 + wave * 16 + r + 8 * hl;
      ctx[(size_t)row * HID + col] = f2bf(cacc[jt][r] * run_sum[r]);
    }
  }
}

// ---------------------------------------------------------------- output projection
// out = ctx @ Wo^T + bo + X (residual), fp32 ; Wo passed pre-transposed
__global__ __launch_bounds__(256)
void oproj_gemm_kernel(const unsigned short* __restrict__ Cbf,  // [S][H]
                       const unsigned short* __restrict__ Wto,  // [H][H] k-major
                       const float* __restrict__ bo,
                       const float* __restrict__ Xres,
                       float* __restrict__ out) {
  __shared__ unsigned short Xs[2][64][32];
  __shared__ unsigned short Ws[2][32][128];

  const int t    = threadIdx.x;
  const int wave = t >> 5, lane = t & 31, lan = lane & 15, hl = lane >> 4;
  const int wm = wave & 3, wn = wave >> 2;
  const int M0 = blockIdx.y * 64;
  const int N0 = blockIdx.x * 128;

  auto issue = [&](int kk, int buf) {
    {
      int row = t >> 2, sub = t & 3;
      ASYNC_CP16(Cbf + (size_t)(M0 + row) * HID + kk * 32 + sub * 8,
                 &Xs[buf][row][sub * 8]);
    }
    #pragma unroll
    for (int i = 0; i < 2; ++i) {
      int c = i * 256 + t;
      int k = c >> 4, sub = c & 15;
      ASYNC_CP16(Wto + (size_t)(kk * 32 + k) * HID + N0 + sub * 8,
                 &Ws[buf][k][sub * 8]);
    }
  };

  v8f zero8 = {0.f,0.f,0.f,0.f,0.f,0.f,0.f,0.f};
  v8f acc[4];
  #pragma unroll
  for (int j = 0; j < 4; ++j) acc[j] = zero8;

  issue(0, 0);
  const int NK = HID / 32;
  for (int kk = 0; kk < NK; ++kk) {
    const int buf = kk & 1;
    if (kk + 1 < NK) { issue(kk + 1, buf ^ 1); STAGE_WAIT(3); }
    else             { STAGE_WAIT(0); }
    __syncthreads();

    Frag af;
    const unsigned int* xrow = (const unsigned int*)&Xs[buf][wm * 16 + lan][0];
    #pragma unroll
    for (int j = 0; j < 8; ++j)
      af.u[j] = xrow[(j & 3) + ((j >> 2) << 3) + (hl << 2)];

    const unsigned int* wsu = (const unsigned int*)&Ws[buf][0][0];
    #pragma unroll
    for (int jt = 0; jt < 4; ++jt) {
      Frag bfr;
      const unsigned short* bcol = &Ws[buf][0][wn * 64 + jt * 16 + lan];
      #pragma unroll
      for (int j = 0; j < 8; ++j) { int k = 2 * (j & 3) + ((j >> 2) << 4) + (hl << 3); bfr.u[j] = bcol[k * 128] | ((unsigned int)bcol[(k + 1) * 128] << 16); }
      acc[jt] = __builtin_amdgcn_wmma_f32_16x16x32_bf16(
          false, af.v, false, bfr.v, (short)0, acc[jt], false, false);
    }
    __syncthreads();
  }

  #pragma unroll
  for (int jt = 0; jt < 4; ++jt) {
    int col = N0 + wn * 64 + jt * 16 + lan;
    float b = bo[col];
    #pragma unroll
    for (int r = 0; r < 8; ++r) {
      int row = M0 + wm * 16 + r + 8 * hl;
      size_t idx = (size_t)row * HID + col;
      out[idx] = acc[jt][r] + b + Xres[idx];
    }
  }
}

// ---------------------------------------------------------------- LayerNorm
__global__ __launch_bounds__(256)
void ln_kernel(const float* __restrict__ x, const float* __restrict__ g,
               const float* __restrict__ b, float* __restrict__ y) {
  __shared__ float ss[8], sqs[8];
  const int t = threadIdx.x, wave = t >> 5, lane = t & 31;
  const int row = blockIdx.x;
  const float4 v = ((const float4*)(x + (size_t)row * HID))[t];
  float s  = v.x + v.y + v.z + v.w;
  float sq = v.x * v.x + v.y * v.y + v.z * v.z + v.w * v.w;
  #pragma unroll
  for (int off = 16; off >= 1; off >>= 1) {
    s  += __shfl_xor(s, off);
    sq += __shfl_xor(sq, off);
  }
  if (lane == 0) { ss[wave] = s; sqs[wave] = sq; }
  __syncthreads();
  float ts = 0.f, tq = 0.f;
  #pragma unroll
  for (int i = 0; i < 8; ++i) { ts += ss[i]; tq += sqs[i]; }
  const float mean = ts * (1.0f / HID);
  const float var  = tq * (1.0f / HID) - mean * mean;
  const float rs   = rsqrtf(var + 1e-12f);
  const float4 gg = ((const float4*)g)[t];
  const float4 bb = ((const float4*)b)[t];
  float4 o;
  o.x = (v.x - mean) * rs * gg.x + bb.x;
  o.y = (v.y - mean) * rs * gg.y + bb.y;
  o.z = (v.z - mean) * rs * gg.z + bb.z;
  o.w = (v.w - mean) * rs * gg.w + bb.w;
  ((float4*)(y + (size_t)row * HID))[t] = o;
}

// ---------------------------------------------------------------- launch
extern "C" void kernel_launch(void* const* d_in, const int* in_sizes, int n_in,
                              void* d_out, int out_size, void* d_ws, size_t ws_size,
                              hipStream_t stream) {
  (void)in_sizes; (void)n_in; (void)out_size; (void)ws_size;
  const float* X    = (const float*)d_in[0];
  const float* mask = (const float*)d_in[1];
  const float* wq   = (const float*)d_in[2];
  const float* bq   = (const float*)d_in[3];
  const float* wk   = (const float*)d_in[4];
  const float* bk   = (const float*)d_in[5];
  const float* wv   = (const float*)d_in[6];
  const float* bv   = (const float*)d_in[7];
  const float* wo   = (const float*)d_in[8];
  const float* bo   = (const float*)d_in[9];
  const float* lng  = (const float*)d_in[10];
  const float* lnb  = (const float*)d_in[11];
  float* out = (float*)d_out;

  char* ws = (char*)d_ws;
  const size_t MB = (size_t)1 << 20;
  unsigned short* Xbf  = (unsigned short*)(ws + 0 * MB);   // 8 MB  [S][H]
  unsigned short* Wqb  = (unsigned short*)(ws + 8 * MB);   // 2 MB  W^T k-major
  unsigned short* Wkb  = (unsigned short*)(ws + 10 * MB);  // 2 MB
  unsigned short* Wvb  = (unsigned short*)(ws + 12 * MB);  // 2 MB
  unsigned short* Wob  = (unsigned short*)(ws + 14 * MB);  // 2 MB
  unsigned short* Qbf  = (unsigned short*)(ws + 16 * MB);  // 8 MB  [NH][S][64]
  unsigned short* Ktb  = (unsigned short*)(ws + 24 * MB);  // 8 MB  [H][S]  (K^T)
  unsigned short* Vbf  = (unsigned short*)(ws + 32 * MB);  // 8 MB  [NH][S][64]
  unsigned short* Ctx  = (unsigned short*)(ws + 40 * MB);  // 8 MB  [S][H]
  float*          Attn = (float*)        (ws + 48 * MB);   // 16 MB [S][H] fp32

  cvt_f32_to_bf16<<<2048, 256, 0, stream>>>(X, Xbf, S_LEN * HID);
  dim3 tgrid(HID / 32, HID / 32), tblk(32, 8);
  cvt_transpose_bf16<<<tgrid, tblk, 0, stream>>>(wq, Wqb);
  cvt_transpose_bf16<<<tgrid, tblk, 0, stream>>>(wk, Wkb);
  cvt_transpose_bf16<<<tgrid, tblk, 0, stream>>>(wv, Wvb);
  cvt_transpose_bf16<<<tgrid, tblk, 0, stream>>>(wo, Wob);

  qkv_gemm_kernel<<<dim3(HID / 128, S_LEN / 64, 3), 256, 0, stream>>>(
      Xbf, Wqb, Wkb, Wvb, bq, bk, bv, Qbf, Ktb, Vbf);

  flash_attn_kernel<<<dim3(S_LEN / 64, NHEAD), 128, 0, stream>>>(
      Qbf, Ktb, Vbf, mask, Ctx);

  oproj_gemm_kernel<<<dim3(HID / 128, S_LEN / 64), 256, 0, stream>>>(
      Ctx, Wob, bo, X, Attn);

  ln_kernel<<<S_LEN, 256, 0, stream>>>(Attn, lng, lnb, out);
}
